// dialted_residual_inception_module_up_74972949119510
// MI455X (gfx1250) — matrix operand order repaired and checked by reading, not compiled, not run on hardware
//
#include <hip/hip_runtime.h>
#include <hip/hip_bf16.h>
#include <math.h>

// MI455X / gfx1250 wave32 WMMA pipeline for deconv + BN + S4D(long-conv-as-
// Toeplitz-matmul) + GLU head. All GEMM stages use v_wmma_f32_16x16x32_f16.
// The Toeplitz operand is served from 16 shift-replicated reversed-kernel
// copies so every lane's diagonal slice is ONE aligned 32B load (no LDS
// gathers, no barriers in the hot kernel).

typedef __attribute__((ext_vector_type(16))) _Float16 v16h;
typedef __attribute__((ext_vector_type(8)))  _Float16 v8h;
typedef __attribute__((ext_vector_type(8)))  float    v8f;

#define Bsz 16
#define Cn  128     // Cin
#define Hh  128     // out channels / d_model
#define Lin 1024
#define Lk  4096    // 4*Lin
#define Nst 64
#define KREV_LEN (Lk + 160)          // 4256, reversed kernel + zero pad
#define NQ (KREV_LEN / 16)           // 266 aligned 16-half chunks

union V16u { v16h v; v8h h[2]; };

#define WMMA_F16(a, b, c) \
  __builtin_amdgcn_wmma_f32_16x16x32_f16(false, (a), false, (b), (short)0, (c), false, false)

// ---------------------------------------------------------------------------
// K0: build f16 staged operands.
//   x16 [b][i][cin]        row-major (A operand of deconv GEMM)
//   w16 [j][cout][cin]     (B operand: per-lane contiguous K run)
//   wo16[o][h]             (B operand of output GEMM, same as w_out layout)
__global__ void k0_prep(const float* __restrict__ x, const float* __restrict__ wdc,
                        const float* __restrict__ wout,
                        _Float16* __restrict__ x16, _Float16* __restrict__ w16,
                        _Float16* __restrict__ wo16) {
  int T = gridDim.x * blockDim.x;
  int g = blockIdx.x * blockDim.x + threadIdx.x;
  for (int i = g; i < Bsz * Cn * Lin; i += T) {           // x: (b, cin, l) flat
    int b = i >> 17, cin = (i >> 10) & 127, l = i & 1023;
    x16[((size_t)((b << 10) + l)) * Cn + cin] = (_Float16)x[i];
  }
  for (int i = g; i < 3 * Cn * Hh; i += T) {              // i = (j*128+cout)*128+cin
    int j = i >> 14, cout = (i >> 7) & 127, cin = i & 127;
    w16[i] = (_Float16)wdc[(cin * Hh + cout) * 3 + j];
  }
  for (int i = g; i < 2 * Hh * Hh; i += T) wo16[i] = (_Float16)wout[i];
}

// ---------------------------------------------------------------------------
// K1: deconv as 3 GEMMs (M=16384 rows (b,i), N=128 cout, K=128 cin) + bias +
// LeakyReLU, scattered into h_pre[b][cout][4i+j]. j==3 lanes are bias-only.
__global__ void k1_deconv(const _Float16* __restrict__ x16, const _Float16* __restrict__ w16,
                          const float* __restrict__ bdc, float* __restrict__ hpre) {
  int tid = threadIdx.x, lane = tid & 31, wv = tid >> 5;
  int rowbase = blockIdx.x * 16;
  int j = blockIdx.y;
  if (j == 3) {
    for (int k = tid; k < 16 * Hh; k += 256) {
      int m = k >> 7, ch = k & 127;
      int row = rowbase + m, b = row >> 10, i = row & 1023;
      float v = bdc[ch]; v = v > 0.f ? v : 0.3f * v;
      hpre[((size_t)(b * Hh + ch)) * Lk + 4 * i + 3] = v;
    }
    return;
  }
  int cout = wv * 16 + (lane & 15);
  v8f acc = {};
  for (int kc = 0; kc < 4; ++kc) {
    int row = rowbase + (lane & 15);
    const _Float16* ab = x16 + (size_t)row * Cn + kc * 32;
    int kb = (lane >> 4) * 16;
    V16u a; a.h[0] = *(const v8h*)(ab + kb); a.h[1] = *(const v8h*)(ab + kb + 8);
    const _Float16* bb = w16 + ((size_t)(j * Hh + cout)) * Cn + kc * 32 + (lane >> 4) * 16;
    v16h bvec = *(const v16h*)bb;
    acc = WMMA_F16(a.v, bvec, acc);
  }
  float bias = bdc[cout];
#pragma unroll
  for (int r = 0; r < 8; ++r) {
    int m = (lane >> 4) * 8 + r;
    int row = rowbase + m, b = row >> 10, i = row & 1023;
    float v = acc[r] + bias;
    v = v > 0.f ? v : 0.3f * v;
    hpre[((size_t)(b * Hh + cout)) * Lk + 4 * i + j] = v;
  }
}

// ---------------------------------------------------------------------------
// K2: per-channel batch stats over (B, Lk); fold gamma/beta into scale/shift.
__global__ void k2_bnstats(const float* __restrict__ hpre, const float* __restrict__ gamma,
                           const float* __restrict__ beta, float* __restrict__ ss) {
  int h = blockIdx.x, tid = threadIdx.x;
  __shared__ float s1[256], s2[256];
  float a = 0.f, q = 0.f;
  for (int i = tid; i < Bsz * Lk; i += 256) {
    int b = i >> 12, l = i & 4095;
    float v = hpre[((size_t)(b * Hh + h)) * Lk + l];
    a += v; q += v * v;
  }
  s1[tid] = a; s2[tid] = q; __syncthreads();
  for (int s = 128; s > 0; s >>= 1) {
    if (tid < s) { s1[tid] += s1[tid + s]; s2[tid] += s2[tid + s]; }
    __syncthreads();
  }
  if (tid == 0) {
    float inv = 1.0f / (float)(Bsz * Lk);
    float m = s1[0] * inv, v = s2[0] * inv - m * m;
    float rstd = rsqrtf(v + 1e-5f);
    float scl = gamma[h] * rstd;
    ss[h] = scl; ss[Hh + h] = beta[h] - m * scl;
  }
}

// ---------------------------------------------------------------------------
// K3: u2[h][b][m] = f16(BN(h_pre)) — A-operand layout for the Toeplitz conv
// (per lane: fixed batch row, contiguous K(m) runs -> aligned v8h loads).
__global__ void k3_buildu(const float* __restrict__ hpre, const float* __restrict__ ss,
                          _Float16* __restrict__ u2) {
  int T = gridDim.x * blockDim.x, g = blockIdx.x * blockDim.x + threadIdx.x;
  for (int i = g; i < Bsz * Hh * Lk; i += T) {
    int m = i & 4095, h = (i >> 12) & 127, b = i >> 19;
    float v = hpre[i] * ss[h] + ss[Hh + h];
    u2[(((size_t)h * Bsz + b) << 12) + m] = (_Float16)v;
  }
}

// ---------------------------------------------------------------------------
// K4: S4D Vandermonde kernel, stored REVERSED with zero pad:
//   krev[h][t] = K[h, Lk-1-t] for t < Lk, else 0   (t in [0, KREV_LEN)).
__global__ void k4_s4dkern(const float* __restrict__ log_dt, const float* __restrict__ log_A_real,
                           const float* __restrict__ A_imag, const float* __restrict__ C_re,
                           const float* __restrict__ C_im, _Float16* __restrict__ krev) {
  int h = blockIdx.x, tid = threadIdx.x;
  __shared__ float ckr[Nst], cki[Nst], dre[Nst], dim_[Nst];
  if (tid < Nst) {
    float dt = __expf(log_dt[h]);
    float ar = -__expf(log_A_real[h * Nst + tid]);
    float ai = A_imag[h * Nst + tid];
    float xr = ar * dt, xi = ai * dt;
    float er = __expf(xr), s, c;
    __sincosf(xi, &s, &c);
    float e1r = er * c - 1.0f, e1i = er * s;            // expm1(dtA)
    float den = ar * ar + ai * ai;
    float rr = (e1r * ar + e1i * ai) / den;             // expm1(dtA)/A
    float ri = (e1i * ar - e1r * ai) / den;
    float cr = C_re[h * Nst + tid], ci = C_im[h * Nst + tid];
    ckr[tid] = cr * rr - ci * ri;
    cki[tid] = cr * ri + ci * rr;
    dre[tid] = xr; dim_[tid] = xi;
  }
  __syncthreads();
  int t = blockIdx.y * 256 + tid;
  if (t >= KREV_LEN) return;
  float val = 0.f;
  if (t < Lk) {
    float l = (float)(Lk - 1 - t), acc = 0.f;
#pragma unroll 4
    for (int n = 0; n < Nst; ++n) {
      float el = __expf(dre[n] * l), s, c;
      __sincosf(dim_[n] * l, &s, &c);
      acc += el * (ckr[n] * c - cki[n] * s);
    }
    val = 2.0f * acc;
  }
  krev[(size_t)h * KREV_LEN + t] = (_Float16)val;
}

// ---------------------------------------------------------------------------
// K4b: 16 shift-replicated aligned copies of krev (B operand table):
//   krevA[s][h][q][i] = krev[h][q*16 + s + i]  -> any base b reads as one
//   aligned v16h from copy s=b&15 at chunk q=b>>4.  ~17 MB, L2-resident.
__global__ void k4b_expand(const _Float16* __restrict__ krev, _Float16* __restrict__ krevA) {
  int T = gridDim.x * blockDim.x, g = blockIdx.x * blockDim.x + threadIdx.x;
  for (int idx = g; idx < 16 * Hh * NQ * 16; idx += T) {
    int i = idx & 15;
    int rest = idx >> 4;
    int q = rest % NQ;
    int rest2 = rest / NQ;
    int h = rest2 & 127, s = rest2 >> 7;
    int t = q * 16 + s + i;
    _Float16 v = (t < KREV_LEN) ? krev[(size_t)h * KREV_LEN + t] : (_Float16)0.0f;
    krevA[idx] = v;
  }
}

// ---------------------------------------------------------------------------
// K5: causal conv as Toeplitz WMMA matmul, LDS-free.  GEMM per h:
//   D[b, l] += A[b, m] * B[m, l],  A = u2 rows (aligned v8h x2),
//   B[m,l] = K[l-m] = krev[Lk-1-l+m]: per lane ONE aligned v16h from krevA.
//   Each wave owns a 16(b) x 16(l) tile; waves independent, per-wave
//   triangular K-bound.  Epilogue: +D*h_bn, GELU, f16 -> yT[(b,l)][h].
__global__ void k5_conv(const _Float16* __restrict__ krevA, const _Float16* __restrict__ u2,
                        const float* __restrict__ hpre, const float* __restrict__ ss,
                        const float* __restrict__ Dp, _Float16* __restrict__ yT) {
  int h = blockIdx.x;
  int lbase = blockIdx.y * 128;
  int tid = threadIdx.x, lane = tid & 31, wv = tid >> 5;
  int lb = lbase + wv * 16;
  int l = lb + (lane & 15);                       // this lane's output column
  float scl = ss[h], sft = ss[Hh + h], Dh = Dp[h];
  v8f acc = {};
  // A stream: row b = lane&15, K-runs at kb16, kb16+8 within each 32-chunk
  const _Float16* abase = u2 + (((size_t)h * Bsz + (lane & 15)) << 12) + (lane >> 4) * 16;
  // B stream: base = Lk-1-l + kb16 (+32 per chunk); shift copy s is invariant
  int base0 = Lk - 1 - l + (lane >> 4) * 16;
  int s = base0 & 15, q0 = base0 >> 4;
  const _Float16* bbase = krevA + (((size_t)(s * Hh + h) * NQ + q0) << 4);
  int nmc = ((lb >> 5) + 4) & ~3;                 // per-wave bound, multiple of 4
#pragma unroll 4
  for (int mc = 0; mc < nmc; ++mc) {
    V16u a;
    a.h[0] = *(const v8h*)(abase + mc * 32);
    a.h[1] = *(const v8h*)(abase + mc * 32 + 8);
    v16h bv = *(const v16h*)(bbase + mc * 32);
    acc = WMMA_F16(a.v, bv, acc);
  }
#pragma unroll
  for (int r = 0; r < 8; ++r) {
    int b = (lane >> 4) * 8 + r;
    float hb = hpre[((size_t)(b * Hh + h)) * Lk + l] * scl + sft;
    float y = acc[r] + Dh * hb;
    float t = y + 0.044715f * y * y * y;          // tanh-GELU (jax default)
    float gel = 0.5f * y * (1.0f + tanhf(0.7978845608f * t));
    yT[(((size_t)b << 12) + (size_t)l) * Hh + h] = (_Float16)gel;
  }
}

// ---------------------------------------------------------------------------
// K6: output GEMM Z[(b,l), o] = Y(65536x128) x Wout^T(128x256) + b_out, then
// GLU fused through LDS (o pairs with o+128), scatter to (B,H,Lk) output.
__global__ void k6_outglu(const _Float16* __restrict__ yT, const _Float16* __restrict__ wo16,
                          const float* __restrict__ bout, float* __restrict__ out) {
  int tid = threadIdx.x, lane = tid & 31, wv = tid >> 5;
  int rowbase = blockIdx.x * 16;
  int o0 = wv * 32 + (lane & 15), o1 = o0 + 16;
  v8f acc0 = {}, acc1 = {};
  for (int kc = 0; kc < 4; ++kc) {
    int row = rowbase + (lane & 15);
    const _Float16* ab = yT + (size_t)row * Hh + kc * 32;
    int kb = (lane >> 4) * 16;
    V16u a; a.h[0] = *(const v8h*)(ab + kb); a.h[1] = *(const v8h*)(ab + kb + 8);
    v16h b0 = *(const v16h*)(wo16 + (size_t)o0 * Hh + kc * 32 + (lane >> 4) * 16);
    v16h b1 = *(const v16h*)(wo16 + (size_t)o1 * Hh + kc * 32 + (lane >> 4) * 16);
    acc0 = WMMA_F16(a.v, b0, acc0);
    acc1 = WMMA_F16(a.v, b1, acc1);
  }
  __shared__ float zl[16][256];
  float bo0 = bout[o0], bo1 = bout[o1];
#pragma unroll
  for (int r = 0; r < 8; ++r) {
    int m = (lane >> 4) * 8 + r;
    zl[m][o0] = acc0[r] + bo0;
    zl[m][o1] = acc1[r] + bo1;
  }
  __syncthreads();
  for (int k = tid; k < 16 * Hh; k += 256) {
    int m = k >> 7, o = k & 127;
    float a_ = zl[m][o], g_ = zl[m][o + 128];
    float val = a_ * (1.0f / (1.0f + __expf(-g_)));
    int grow = rowbase + m;
    int b = grow >> 12, l = grow & 4095;
    out[((size_t)(b * Hh + o)) * Lk + l] = val;
  }
}

// ---------------------------------------------------------------------------
extern "C" void kernel_launch(void* const* d_in, const int* in_sizes, int n_in,
                              void* d_out, int out_size, void* d_ws, size_t ws_size,
                              hipStream_t stream) {
  (void)in_sizes; (void)n_in; (void)out_size; (void)ws_size;
  const float* x          = (const float*)d_in[0];
  const float* wdc        = (const float*)d_in[1];
  const float* bdc        = (const float*)d_in[2];
  const float* gamma      = (const float*)d_in[3];
  const float* beta       = (const float*)d_in[4];
  const float* log_dt     = (const float*)d_in[5];
  const float* log_A_real = (const float*)d_in[6];
  const float* A_imag     = (const float*)d_in[7];
  const float* C_re       = (const float*)d_in[8];
  const float* C_im       = (const float*)d_in[9];
  const float* Dp         = (const float*)d_in[10];
  const float* wout       = (const float*)d_in[11];
  const float* bout       = (const float*)d_in[12];
  float* out = (float*)d_out;

  char* ws = (char*)d_ws;
  size_t off = 0;
  auto alloc = [&](size_t bytes) -> void* {
    void* p = ws + off;
    off += (bytes + 255) & ~(size_t)255;
    return p;
  };
  float*     hpre  = (float*)    alloc(sizeof(float)    * Bsz * Hh * Lk);       // 33.5 MB
  _Float16*  x16   = (_Float16*) alloc(sizeof(_Float16) * Bsz * Lin * Cn);      //  4 MB
  _Float16*  w16   = (_Float16*) alloc(sizeof(_Float16) * 3 * Hh * Cn);
  _Float16*  wo16  = (_Float16*) alloc(sizeof(_Float16) * 2 * Hh * Hh);
  _Float16*  u2    = (_Float16*) alloc(sizeof(_Float16) * Hh * Bsz * Lk);       // 16 MB
  _Float16*  krev  = (_Float16*) alloc(sizeof(_Float16) * Hh * KREV_LEN);       //  1.1 MB
  _Float16*  krevA = (_Float16*) alloc(sizeof(_Float16) * 16 * Hh * NQ * 16);   // 17.4 MB
  _Float16*  yT    = (_Float16*) alloc(sizeof(_Float16) * Bsz * Lk * Hh);       // 16 MB
  float*     ss    = (float*)    alloc(sizeof(float) * 2 * Hh);

  k0_prep   <<<2048, 256, 0, stream>>>(x, wdc, wout, x16, w16, wo16);
  k1_deconv <<<dim3(1024, 4, 1), 256, 0, stream>>>(x16, w16, bdc, hpre);
  k2_bnstats<<<128, 256, 0, stream>>>(hpre, gamma, beta, ss);
  k3_buildu <<<8192, 256, 0, stream>>>(hpre, ss, u2);
  k4_s4dkern<<<dim3(128, 17, 1), 256, 0, stream>>>(log_dt, log_A_real, A_imag, C_re, C_im, krev);
  k4b_expand<<<4096, 256, 0, stream>>>(krev, krevA);
  k5_conv   <<<dim3(128, 32, 1), 256, 0, stream>>>(krevA, u2, hpre, ss, Dp, yT);
  k6_outglu <<<4096, 256, 0, stream>>>(yT, wo16, bout, out);
}
